// MStAAttention_70334384439336
// MI455X (gfx1250) — matrix operand order repaired and checked by reading, not compiled, not run on hardware
//
#include <hip/hip_runtime.h>

typedef __attribute__((ext_vector_type(16))) _Float16 v16h;
typedef __attribute__((ext_vector_type(8)))  _Float16 v8h;
typedef __attribute__((ext_vector_type(8)))  float    v8f;

#define D_DIM 256
#define S_DIM 17
#define H_HEADS 8
#define C_DIM 32

// ---------------------------------------------------------------------------
// Scan: rank-matched scatter mapping. For each flat dst row j in [0, M*N):
//   row_src[j] = flat index into h[1:] rows that lands there, or -1.
// Single 1024-thread block, Hillis-Steele scan in LDS.
// ---------------------------------------------------------------------------
__global__ void scan_kernel(const unsigned char* __restrict__ src,
                            const unsigned char* __restrict__ dst,
                            int* __restrict__ src_pos,
                            int* __restrict__ row_src, int MN) {
  __shared__ int sS[1024];
  __shared__ int sD[1024];
  int t = threadIdx.x;
  int per = (MN + 1023) >> 10;
  int base = t * per;
  int cs = 0, cd = 0;
  for (int i = 0; i < per; ++i) {
    int j = base + i;
    if (j < MN) { cs += src[j] ? 1 : 0; cd += dst[j] ? 1 : 0; }
  }
  sS[t] = cs; sD[t] = cd;
  __syncthreads();
  for (int off = 1; off < 1024; off <<= 1) {
    int a = (t >= off) ? sS[t - off] : 0;
    int b = (t >= off) ? sD[t - off] : 0;
    __syncthreads();
    sS[t] += a; sD[t] += b;
    __syncthreads();
  }
  int exS = sS[t] - cs;
  int exD = sD[t] - cd;
  int r = exS;
  for (int i = 0; i < per; ++i) {
    int j = base + i;
    if (j < MN && src[j]) src_pos[r++] = j;
  }
  __threadfence();
  __syncthreads();
  int rd = exD;
  for (int i = 0; i < per; ++i) {
    int j = base + i;
    if (j < MN) {
      if (dst[j]) {
        int rank = rd < MN ? rd : MN - 1;
        row_src[j] = src_pos[rank];
        rd++;
      } else {
        row_src[j] = -1;
      }
    }
  }
}

// ---------------------------------------------------------------------------
// f32 -> f16 convert
// ---------------------------------------------------------------------------
__global__ void cvt_f32_f16(const float* __restrict__ in, _Float16* __restrict__ out, size_t n) {
  size_t i = (size_t)blockIdx.x * blockDim.x + threadIdx.x;
  if (i < n) out[i] = (_Float16)in[i];
}

// ---------------------------------------------------------------------------
// Build attn_h (f16): row 0 = h[0]; rows s>=1 gathered via row_src or zero.
// ---------------------------------------------------------------------------
__global__ void build_a16(const float* __restrict__ h, const int* __restrict__ row_src,
                          _Float16* __restrict__ A16, int N, int MN) {
  size_t i = (size_t)blockIdx.x * blockDim.x + threadIdx.x;
  size_t total = (size_t)S_DIM * N * D_DIM;
  if (i >= total) return;
  int d = (int)(i % D_DIM);
  size_t row = i / D_DIM;   // s*N + n
  float val;
  if (row < (size_t)N) {
    val = h[i];             // s == 0
  } else {
    size_t j = row - (size_t)N;   // (s-1)*N + n
    int rs = row_src[j];
    if (rs < 0) {
      val = 0.0f;
    } else {
      if (rs > MN - 1) rs = MN - 1;
      val = h[((size_t)N + (size_t)rs) * D_DIM + d];   // h[1:] flat row rs
    }
  }
  A16[i] = (_Float16)val;
}

// ---------------------------------------------------------------------------
// WMMA fragment loads. Row-major matrices, ld = 256 halves.
// A frag (16x32 f16): lane = (M = lane&15, hi = lane>>4);
//   halves [0..15] = K in [16*hi, 16*hi+16)
// B frag (32x16 f16), B[kk][nn] = W[n0+nn][k0+kk] (out = in @ W^T):
//   lane = (N = lane&15, hi = lane>>4); halves [0..15] = K in [16*hi, 16*hi+16)
// ---------------------------------------------------------------------------
__device__ __forceinline__ v16h load_frag_A(const _Float16* __restrict__ base,
                                            int row0, int k0, int lane) {
  int r   = row0 + (lane & 15);
  int hi8 = (lane >> 4) << 4;
  const _Float16* p = base + (size_t)r * D_DIM + k0 + hi8;
  v8h lo = *(const v8h*)(p);
  v8h hh = *(const v8h*)(p + 8);
  return __builtin_shufflevector(lo, hh, 0,1,2,3,4,5,6,7,8,9,10,11,12,13,14,15);
}

__device__ __forceinline__ v16h load_frag_B(const _Float16* __restrict__ W,
                                            int n0, int k0, int lane) {
  int r    = n0 + (lane & 15);
  int hi16 = (lane >> 4) << 4;
  return *(const v16h*)(W + (size_t)r * D_DIM + k0 + hi16);
}

// ---------------------------------------------------------------------------
// GEMM: out[R x 256] (f16) = A[R x 256] (f16) @ W^T  (W is 256x256 row-major)
// One wave computes a 32x64 tile: 2x4 accumulators, K looped in 8 chunks of 32.
// ---------------------------------------------------------------------------
__global__ void gemm_qkv(const _Float16* __restrict__ A, const _Float16* __restrict__ W,
                         _Float16* __restrict__ out, int R) {
  int lane  = threadIdx.x & 31;
  int wid   = blockIdx.x * (blockDim.x >> 5) + (threadIdx.x >> 5);
  int tiles = (R >> 5) * 4;           // (R/32) row tiles x 4 col groups of 64
  if (wid >= tiles) return;
  int m0 = (wid >> 2) << 5;
  int n0 = (wid & 3) << 6;
  v8f c[2][4] = {};
#pragma unroll
  for (int kk = 0; kk < 8; ++kk) {
    int k0 = kk << 5;
    v16h a0 = load_frag_A(A, m0,      k0, lane);
    v16h a1 = load_frag_A(A, m0 + 16, k0, lane);
#pragma unroll
    for (int j = 0; j < 4; ++j) {
      v16h b = load_frag_B(W, n0 + 16 * j, k0, lane);
      c[0][j] = __builtin_amdgcn_wmma_f32_16x16x32_f16(false, a0, false, b, (short)0, c[0][j], false, false);
      c[1][j] = __builtin_amdgcn_wmma_f32_16x16x32_f16(false, a1, false, b, (short)0, c[1][j], false, false);
    }
  }
  int nn = lane & 15, hi = lane >> 4;
#pragma unroll
  for (int i = 0; i < 2; ++i)
#pragma unroll
    for (int j = 0; j < 4; ++j)
#pragma unroll
      for (int r = 0; r < 8; ++r) {
        int row = m0 + 16 * i + r + 8 * hi;
        int col = n0 + 16 * j + nn;
        out[(size_t)row * D_DIM + col] = (_Float16)c[i][j][r];
      }
}

// ---------------------------------------------------------------------------
// Final GEMM with epilogue: y = o16 @ Wo^T + bo + h   (f32 out, pre-LayerNorm)
// ---------------------------------------------------------------------------
__global__ void gemm_out(const _Float16* __restrict__ A, const _Float16* __restrict__ W,
                         const float* __restrict__ bo, const float* __restrict__ hres,
                         float* __restrict__ y, int R) {
  int lane  = threadIdx.x & 31;
  int wid   = blockIdx.x * (blockDim.x >> 5) + (threadIdx.x >> 5);
  int tiles = (R >> 5) * 4;
  if (wid >= tiles) return;
  int m0 = (wid >> 2) << 5;
  int n0 = (wid & 3) << 6;
  v8f c[2][4] = {};
#pragma unroll
  for (int kk = 0; kk < 8; ++kk) {
    int k0 = kk << 5;
    v16h a0 = load_frag_A(A, m0,      k0, lane);
    v16h a1 = load_frag_A(A, m0 + 16, k0, lane);
#pragma unroll
    for (int j = 0; j < 4; ++j) {
      v16h b = load_frag_B(W, n0 + 16 * j, k0, lane);
      c[0][j] = __builtin_amdgcn_wmma_f32_16x16x32_f16(false, a0, false, b, (short)0, c[0][j], false, false);
      c[1][j] = __builtin_amdgcn_wmma_f32_16x16x32_f16(false, a1, false, b, (short)0, c[1][j], false, false);
    }
  }
  int nn = lane & 15, hi = lane >> 4;
#pragma unroll
  for (int i = 0; i < 2; ++i)
#pragma unroll
    for (int j = 0; j < 4; ++j) {
      int col = n0 + 16 * j + nn;
      float bias = bo[col];
#pragma unroll
      for (int r = 0; r < 8; ++r) {
        int row = m0 + 16 * i + r + 8 * hi;
        size_t off = (size_t)row * D_DIM + col;
        y[off] = c[i][j][r] + bias + hres[off];
      }
    }
}

// ---------------------------------------------------------------------------
// Attention: one thread per (s, head, n). 17x17 scores over C=32, masked
// softmax over key axis t, weighted sum of v, then select attn_o vs hv.
// ---------------------------------------------------------------------------
__global__ void attn_kernel(const _Float16* __restrict__ q16, const _Float16* __restrict__ k16,
                            const _Float16* __restrict__ v16, const _Float16* __restrict__ hv16,
                            const unsigned char* __restrict__ tmask,
                            _Float16* __restrict__ o16, int N) {
  int idx = blockIdx.x * blockDim.x + threadIdx.x;
  int total = S_DIM * H_HEADS * N;
  if (idx >= total) return;
  int n = idx % N;
  int tmp = idx / N;
  int head = tmp % H_HEADS;
  int s = tmp / H_HEADS;

  bool colsn = false;
#pragma unroll
  for (int t = 0; t < S_DIM - 1; ++t) colsn = colsn || (tmask[(size_t)t * N + n] != 0);

  const float scale = 0.17677669529663687f;   // 1/sqrt(32)
  float qv[C_DIM];
  {
    const v8h* qp = (const v8h*)(q16 + ((size_t)s * N + n) * D_DIM + head * C_DIM);
    v8h qc[4] = {qp[0], qp[1], qp[2], qp[3]};
#pragma unroll
    for (int c = 0; c < C_DIM; ++c) qv[c] = (float)qc[c >> 3][c & 7];
  }

  float sc[S_DIM];
  float mx = -3.4e38f;
  for (int t = 0; t < S_DIM; ++t) {
    const v8h* kp = (const v8h*)(k16 + ((size_t)t * N + n) * D_DIM + head * C_DIM);
    v8h kc[4] = {kp[0], kp[1], kp[2], kp[3]};
    float d = 0.0f;
#pragma unroll
    for (int c = 0; c < C_DIM; ++c) d += qv[c] * (float)kc[c >> 3][c & 7];
    bool km = (t == 0) ? colsn : (tmask[(size_t)(t - 1) * N + n] != 0);
    d = d * scale + (km ? 0.0f : -1.0e9f);
    sc[t] = d;
    mx = fmaxf(mx, d);
  }
  float sum = 0.0f;
#pragma unroll
  for (int t = 0; t < S_DIM; ++t) { sc[t] = __expf(sc[t] - mx); sum += sc[t]; }
  float inv = 1.0f / sum;

  float acc[C_DIM];
#pragma unroll
  for (int c = 0; c < C_DIM; ++c) acc[c] = 0.0f;
  for (int t = 0; t < S_DIM; ++t) {
    float p = sc[t] * inv;
    const v8h* vp = (const v8h*)(v16 + ((size_t)t * N + n) * D_DIM + head * C_DIM);
    v8h vc[4] = {vp[0], vp[1], vp[2], vp[3]};
#pragma unroll
    for (int c = 0; c < C_DIM; ++c) acc[c] += p * (float)vc[c >> 3][c & 7];
  }

  bool qm = (s == 0) ? colsn : (tmask[(size_t)(s - 1) * N + n] != 0);
  size_t off = ((size_t)s * N + n) * D_DIM + head * C_DIM;
  _Float16* op = o16 + off;
  const _Float16* hp = hv16 + off;
#pragma unroll
  for (int c = 0; c < C_DIM; ++c) op[c] = qm ? (_Float16)acc[c] : hp[c];
}

// ---------------------------------------------------------------------------
// LayerNorm: one wave per row of 256 (8 elems/lane), shfl_xor reductions.
// ---------------------------------------------------------------------------
__device__ __forceinline__ float wave_sum(float v) {
#pragma unroll
  for (int off = 16; off > 0; off >>= 1) v += __shfl_xor(v, off, 32);
  return v;
}

__global__ void ln_kernel(const float* __restrict__ y, const float* __restrict__ g,
                          const float* __restrict__ b, float* __restrict__ out, int R) {
  int wid = blockIdx.x * (blockDim.x >> 5) + (threadIdx.x >> 5);
  if (wid >= R) return;
  int lane = threadIdx.x & 31;
  const float* yr = y + (size_t)wid * D_DIM;
  float v[8];
  float s = 0.0f;
#pragma unroll
  for (int i = 0; i < 8; ++i) { v[i] = yr[lane * 8 + i]; s += v[i]; }
  float mu = wave_sum(s) * (1.0f / 256.0f);
  float q = 0.0f;
#pragma unroll
  for (int i = 0; i < 8; ++i) { float d = v[i] - mu; q += d * d; }
  float var = wave_sum(q) * (1.0f / 256.0f);
  float inv = rsqrtf(var + 1e-5f);
  float* orow = out + (size_t)wid * D_DIM;
#pragma unroll
  for (int i = 0; i < 8; ++i) {
    int c = lane * 8 + i;
    orow[c] = (v[i] - mu) * inv * g[c] + b[c];
  }
}

// ---------------------------------------------------------------------------
// Launch
// ---------------------------------------------------------------------------
extern "C" void kernel_launch(void* const* d_in, const int* in_sizes, int n_in,
                              void* d_out, int out_size, void* d_ws, size_t ws_size,
                              hipStream_t stream) {
  const float* h            = (const float*)d_in[0];
  const unsigned char* tmsk = (const unsigned char*)d_in[1];  // target_msta_mask [16,N]
  const unsigned char* smsk = (const unsigned char*)d_in[2];  // msta_mask        [16,N]
  const float* Wq = (const float*)d_in[3];
  const float* Wk = (const float*)d_in[4];
  const float* Wv = (const float*)d_in[5];
  const float* Wo = (const float*)d_in[6];
  const float* bo = (const float*)d_in[7];
  const float* lg = (const float*)d_in[8];
  const float* lb = (const float*)d_in[9];
  float* out = (float*)d_out;

  const int N  = in_sizes[1] / (S_DIM - 1);     // 4096
  const int MN = (S_DIM - 1) * N;
  const int R  = S_DIM * N;                     // 69632 rows
  const size_t RD = (size_t)R * D_DIM;

  char* w = (char*)d_ws;
  auto alloc = [&](size_t bytes) -> void* {
    void* p = (void*)w;
    w += (bytes + 255) & ~(size_t)255;
    return p;
  };
  _Float16* A16  = (_Float16*)alloc(RD * 2);
  _Float16* H16  = (_Float16*)alloc(RD * 2);
  _Float16* Wq16 = (_Float16*)alloc((size_t)D_DIM * D_DIM * 2);
  _Float16* Wk16 = (_Float16*)alloc((size_t)D_DIM * D_DIM * 2);
  _Float16* Wv16 = (_Float16*)alloc((size_t)D_DIM * D_DIM * 2);
  _Float16* Wo16 = (_Float16*)alloc((size_t)D_DIM * D_DIM * 2);
  _Float16* q16  = (_Float16*)alloc(RD * 2);
  _Float16* k16  = (_Float16*)alloc(RD * 2);
  _Float16* v16  = (_Float16*)alloc(RD * 2);
  _Float16* hv16 = (_Float16*)alloc(RD * 2);
  _Float16* o16  = (_Float16*)alloc(RD * 2);
  float*    ybuf = (float*)alloc(RD * 4);
  int* src_pos   = (int*)alloc((size_t)MN * 4);
  int* row_src   = (int*)alloc((size_t)MN * 4);

  // 1) scatter mapping
  scan_kernel<<<1, 1024, 0, stream>>>(smsk, tmsk, src_pos, row_src, MN);

  // 2) pack f16 operands
  {
    int blocks = (int)((RD + 255) / 256);
    cvt_f32_f16<<<blocks, 256, 0, stream>>>(h, H16, RD);
    build_a16<<<blocks, 256, 0, stream>>>(h, row_src, A16, N, MN);
    int wb = (D_DIM * D_DIM + 255) / 256;
    cvt_f32_f16<<<wb, 256, 0, stream>>>(Wq, Wq16, (size_t)D_DIM * D_DIM);
    cvt_f32_f16<<<wb, 256, 0, stream>>>(Wk, Wk16, (size_t)D_DIM * D_DIM);
    cvt_f32_f16<<<wb, 256, 0, stream>>>(Wv, Wv16, (size_t)D_DIM * D_DIM);
    cvt_f32_f16<<<wb, 256, 0, stream>>>(Wo, Wo16, (size_t)D_DIM * D_DIM);
  }

  // 3) projections (WMMA f16): q,k,v from attn_h; fallback hv from h
  int tiles  = (R >> 5) * 4;
  int gblk   = (tiles + 7) / 8;      // 8 waves per 256-thread block
  gemm_qkv<<<gblk, 256, 0, stream>>>(A16, Wq16, q16, R);
  gemm_qkv<<<gblk, 256, 0, stream>>>(A16, Wk16, k16, R);
  gemm_qkv<<<gblk, 256, 0, stream>>>(A16, Wv16, v16, R);
  gemm_qkv<<<gblk, 256, 0, stream>>>(H16, Wv16, hv16, R);

  // 4) masked softmax attention + select
  {
    int total  = S_DIM * H_HEADS * N;
    int blocks = (total + 255) / 256;
    attn_kernel<<<blocks, 256, 0, stream>>>(q16, k16, v16, hv16, tmsk, o16, N);
  }

  // 5) output projection + bias + residual (WMMA f16, f32 epilogue)
  gemm_out<<<gblk, 256, 0, stream>>>(o16, Wo16, bo, h, ybuf, R);

  // 6) LayerNorm -> d_out
  {
    int blocks = (R + 7) / 8;
    ln_kernel<<<blocks, 256, 0, stream>>>(ybuf, lg, lb, out, R);
  }
  (void)n_in; (void)out_size; (void)ws_size;
}
